// UltimateNeuralLayer_50964081934819
// MI455X (gfx1250) — compile-verified, matrix operand order checked by reading, not run on hardware
//
#include <hip/hip_runtime.h>
#include <math.h>
#include <stdint.h>

// ---- problem dims (fixed by the reference) ----
#define B_DIM 16384
#define N_DIM 4096
#define J_DIM 256
#define D_DIM 9

// ---- CDNA5 WMMA vector types ----
typedef __attribute__((ext_vector_type(16))) _Float16 v16h;
typedef __attribute__((ext_vector_type(8)))  float    v8f;

// ---- GEMM tiling ----
#define MBLK 128
#define NBLK 128
#define KC   64
#define LDS_STRIDE 80                    // halves: 160B row stride -> 32B-aligned frags
#define TILE_HALVES (128 * LDS_STRIDE)   // one tile (X or W), one buffer

// CDNA5 async memory->LDS copy (GV mode: 64-bit VGPR address, no saddr).
// Tracked by ASYNCcnt; published to the workgroup via s_wait_asynccnt + barrier.
static __device__ __forceinline__ void asyncLoadB128(unsigned ldsAddr,
                                                     unsigned long long gAddr) {
    asm volatile("global_load_async_to_lds_b128 %0, %1, off"
                 :: "v"(ldsAddr), "v"(gAddr) : "memory");
}
static __device__ __forceinline__ void waitAsync0() {
#if __has_builtin(__builtin_amdgcn_s_wait_asynccnt)
    __builtin_amdgcn_s_wait_asynccnt(0);
#else
    asm volatile("s_wait_asynccnt 0x0" ::: "memory");
#endif
}
static __device__ __forceinline__ void schedBarrier() {
#if __has_builtin(__builtin_amdgcn_sched_barrier)
    __builtin_amdgcn_sched_barrier(0);
#endif
}

// Pack two f32 into two f16 in ONE VALU op (v_cvt_pk_rtz_f16_f32).
static __device__ __forceinline__ unsigned pkF16(float a, float b) {
    auto h = __builtin_amdgcn_cvt_pkrtz(a, b);
    return __builtin_bit_cast(unsigned, h);
}

// ---------------------------------------------------------------------------
// Kernel 1: closed-form "quantum" probability + bias fuse.
// state starts constant along axis 1; the roll terms of _rotate cancel on a
// constant row, so state[j,:] = (1/sqrt(N)) * prod cos(angles[:,j,:]) and
// prob[j] = prod(cos)^2 / N.   probplus[j] = prob[j] + bias[j].
// ---------------------------------------------------------------------------
__global__ void unl_prob_kernel(const float* __restrict__ uw,   // (D, N, J)
                                const float* __restrict__ bias, // (J)
                                float* __restrict__ probplus)   // (J)
{
    int j = threadIdx.x;            // 256 threads, one per output column
    float p = 1.0f;
    #pragma unroll
    for (int d = 0; d < D_DIM; ++d) {
        const float* a = uw + (size_t)d * N_DIM * J_DIM + (size_t)j * J_DIM;
        p *= cosf(a[0]);
        p *= cosf(a[1]);
        p *= cosf(a[2]);
    }
    probplus[j] = p * p * (1.0f / (float)N_DIM) + bias[j];
}

// ---------------------------------------------------------------------------
// Kernel 2: W (4096 x 256, f32, row-major) -> Wt (256 x 4096, f16, row-major)
// LDS-tiled transpose so both global read and write are coalesced.
// ---------------------------------------------------------------------------
__global__ void unl_wt_kernel(const float* __restrict__ W,
                              _Float16* __restrict__ Wt)
{
    __shared__ float tile[16][17];
    const int jBase = blockIdx.x * 16;   // over J (256)
    const int kBase = blockIdx.y * 16;   // over N (4096)
    const int tx = threadIdx.x, ty = threadIdx.y;
    tile[ty][tx] = W[(size_t)(kBase + ty) * J_DIM + (jBase + tx)];
    __syncthreads();
    Wt[(size_t)(jBase + ty) * N_DIM + (kBase + tx)] = (_Float16)tile[tx][ty];
}

// ---------------------------------------------------------------------------
// Kernel 3: out[b,j] = tanh( x[b,:] . W[:,j] + probplus[j] )
// f16 WMMA GEMM, f32 accumulate. 256 threads = 8 waves, 128x128 tile/block.
// Double-buffered LDS: Wt tiles via global_load_async_to_lds_b128 (ASYNCcnt),
// x tiles via f32 vector loads + v_cvt_pk_rtz_f16_f32 + ds_store, both
// overlapped with the WMMA chain of the current buffer. A sched_barrier pins
// the batched fragment loads ahead of the 8 back-to-back WMMAs.
// ---------------------------------------------------------------------------
__global__ __launch_bounds__(256) void unl_gemm_kernel(
    const float*    __restrict__ x,        // (B, N) f32
    const _Float16* __restrict__ Wt,       // (J, N) f16
    const float*    __restrict__ probplus, // (J)
    float*          __restrict__ out)      // (B, J)
{
    // dynamic LDS: [X buf0 | X buf1 | W buf0 | W buf1], each TILE_HALVES
    extern __shared__ _Float16 smem[];

    const int tid  = threadIdx.x;
    const int wave = tid >> 5;            // 0..7 (wave32)
    const int lane = tid & 31;
    const int rowBase = blockIdx.y * MBLK;
    const int colBase = blockIdx.x * NBLK;

    v8f acc[8] = {};                      // 8 column tiles of 16x16 f32

    // cooperative load mapping: 2 threads per row, 32 elements each
    const int ldRow  = tid >> 1;          // 0..127
    const int ldHalf = (tid & 1) * 32;    // 0 or 32
    const int ldsOff = ldRow * LDS_STRIDE + ldHalf;

    const float*    xSrc = x  + (size_t)(rowBase + ldRow) * N_DIM + ldHalf;
    const _Float16* wSrc = Wt + (size_t)(colBase + ldRow) * N_DIM + ldHalf;

    // WMMA fragment addressing: lane -> row = lane&15, K chunk = (lane>>4)*16
    const int fRow = lane & 15;
    const int fK   = (lane >> 4) * 16;

    auto loadW = [&](int buf, int k0) {   // async memory -> LDS, 64 B/thread
        unsigned ldsA = (unsigned)(uintptr_t)&smem[(2 + buf) * TILE_HALVES + ldsOff];
        unsigned long long g = (unsigned long long)(uintptr_t)(wSrc + k0);
        #pragma unroll
        for (int v = 0; v < 4; ++v)
            asyncLoadB128(ldsA + (unsigned)(v * 16), g + (unsigned)(v * 16));
    };
    auto loadX = [&](float4* xr, int k0) {
        const float4* xs = reinterpret_cast<const float4*>(xSrc + k0);
        #pragma unroll
        for (int v = 0; v < 8; ++v) xr[v] = xs[v];
    };
    auto storeX = [&](int buf, const float4* xr) {
        // packed 2xf16 words -> guaranteed ds_store_b128 merging
        unsigned* d32 = reinterpret_cast<unsigned*>(&smem[buf * TILE_HALVES + ldsOff]);
        #pragma unroll
        for (int v = 0; v < 8; ++v) {
            d32[v * 2 + 0] = pkF16(xr[v].x, xr[v].y);
            d32[v * 2 + 1] = pkF16(xr[v].z, xr[v].w);
        }
    };
    auto compute = [&](int buf) {
        const _Float16* xT = &smem[buf * TILE_HALVES];
        const _Float16* wT = &smem[(2 + buf) * TILE_HALVES];
        #pragma unroll
        for (int kk = 0; kk < KC; kk += 32) {
            // batch-load A fragment + all 8 B fragments ...
            const v16h a = *reinterpret_cast<const v16h*>(
                &xT[(wave * 16 + fRow) * LDS_STRIDE + kk + fK]);
            v16h b[8];
            #pragma unroll
            for (int nt = 0; nt < 8; ++nt)
                b[nt] = *reinterpret_cast<const v16h*>(
                    &wT[(nt * 16 + fRow) * LDS_STRIDE + kk + fK]);
            schedBarrier();   // ... keep loads ahead of the WMMA group
            #pragma unroll
            for (int nt = 0; nt < 8; ++nt)
                acc[nt] = __builtin_amdgcn_wmma_f32_16x16x32_f16(
                    /*neg_a=*/false, a, /*neg_b=*/false, b[nt],
                    /*c_mod=*/(short)0, acc[nt],
                    /*reuse_a=*/false, /*reuse_b=*/false);
        }
    };

    // ---- prologue: fill buffer 0 ----
    {
        loadW(0, 0);
        float4 xr[8];
        loadX(xr, 0);
        storeX(0, xr);
        waitAsync0();
    }
    __syncthreads();

    // ---- ping-pong main loop: one barrier per K-step ----
    int cur = 0;
    for (int k0 = 0; k0 < N_DIM; k0 += KC) {
        const bool hasNext = (k0 + KC) < N_DIM;
        float4 xr[8];
        if (hasNext) {
            loadW(cur ^ 1, k0 + KC);          // async -> LDS (next buffer)
            loadX(xr, k0 + KC);               // f32 -> VGPRs (next buffer)
            if (k0 + 2 * KC < N_DIM)          // prefetch the tile after next
                __builtin_prefetch(xSrc + k0 + 2 * KC, 0, 3);
        }
        compute(cur);                          // WMMA chain overlaps the loads
        if (hasNext) {
            storeX(cur ^ 1, xr);              // packed cvt + ds_store into next buffer
            waitAsync0();                      // next W tile landed in LDS
        }
        __syncthreads();                       // publish next buf / retire cur
        cur ^= 1;
    }

    // ---- epilogue ----
    // C/D layout: VGPR r -> M = r (lanes 0-15) / r+8 (lanes 16-31), N = lane&15
    const int outRow0 = rowBase + wave * 16 + ((lane >> 4) * 8);
    const int nCol    = lane & 15;
    #pragma unroll
    for (int nt = 0; nt < 8; ++nt) {
        const int col = colBase + nt * 16 + nCol;
        const float addv = probplus[col];
        #pragma unroll
        for (int r = 0; r < 8; ++r) {
            out[(size_t)(outRow0 + r) * J_DIM + col] = tanhf(acc[nt][r] + addv);
        }
    }
}

// ---------------------------------------------------------------------------
extern "C" void kernel_launch(void* const* d_in, const int* in_sizes, int n_in,
                              void* d_out, int out_size, void* d_ws, size_t ws_size,
                              hipStream_t stream) {
    const float* x    = (const float*)d_in[0];  // (16384, 4096)
    const float* uw   = (const float*)d_in[1];  // (9, 4096, 256)
    const float* W    = (const float*)d_in[2];  // (4096, 256)
    const float* bias = (const float*)d_in[3];  // (256)
    float* out = (float*)d_out;                 // (16384, 256)

    // workspace: [0,1KB) probplus (256 f32), [1KB, 1KB+2MB) Wt f16
    float*    probplus = (float*)d_ws;
    _Float16* Wt       = (_Float16*)((char*)d_ws + 1024);

    constexpr size_t SMEM_BYTES = (size_t)4 * TILE_HALVES * sizeof(_Float16); // 80 KB
    // gfx1250 allows up to 320 KB LDS per workgroup; raise the dynamic cap.
    (void)hipFuncSetAttribute((const void*)unl_gemm_kernel,
                              hipFuncAttributeMaxDynamicSharedMemorySize,
                              (int)SMEM_BYTES);

    unl_prob_kernel<<<1, J_DIM, 0, stream>>>(uw, bias, probplus);
    unl_wt_kernel<<<dim3(J_DIM / 16, N_DIM / 16), dim3(16, 16), 0, stream>>>(W, Wt);
    unl_gemm_kernel<<<dim3(J_DIM / NBLK, B_DIM / MBLK), 256, SMEM_BYTES, stream>>>(
        x, Wt, probplus, out);
}
